// MPSN_29257317220558
// MI455X (gfx1250) — matrix operand order repaired and checked by reading, not compiled, not run on hardware
//
#include <hip/hip_runtime.h>
#include <math.h>

// ---------------------------------------------------------------------------
// Types for CDNA5 WMMA (wave32, 16x16x32 bf16 -> f32 accum)
// ---------------------------------------------------------------------------
typedef __attribute__((ext_vector_type(16))) __bf16 v16bf;
typedef __attribute__((ext_vector_type(8)))  float  v8f;

union BF16x16 {
    v16bf v;
    unsigned short u[16];
};

__device__ __forceinline__ unsigned short f2bf(float f) {
    // round-to-nearest-even f32 -> bf16
    unsigned u = __float_as_uint(f);
    unsigned r = (u + 0x7FFFu + ((u >> 16) & 1u)) >> 16;
    return (unsigned short)r;
}

__device__ __forceinline__ unsigned pack2bf(float lo, float hi) {
    return (unsigned)f2bf(lo) | ((unsigned)f2bf(hi) << 16);
}

// ---------------------------------------------------------------------------
// Tiled GEMM: D = act( (ACC ? C : 0) + opA(A)[MxK] @ opB(B)[KxN] )
//   f32 in HBM, bf16 in LDS, f32 WMMA accumulation.
// Requires M%128==0, N%128==0, K%32==0 (true for every GEMM in this model).
// Block: 256 threads (8 waves). Tile BM=128, BN=128, BK=32.
// Wave w: (w&3) -> 32-row group, (w>>2) -> 64-col group => 2x4 WMMA tiles.
// ---------------------------------------------------------------------------
#define BM 128
#define BN 128
#define BK 32
#define LDSS 34   // padded ushort stride (68B = 17 banks)

template <int TRANSA, int TRANSB, int ACC, int ACT>
__global__ __launch_bounds__(256)
void gemm_wmma_bf16(const float* __restrict__ A, const float* __restrict__ B,
                    const float* __restrict__ C, float* __restrict__ D,
                    int M, int N, int K, int lda, int ldb, int ldc)
{
    __shared__ unsigned short As[BM * LDSS];
    __shared__ unsigned short Bs[BN * LDSS];

    const int t    = threadIdx.x;
    const int lane = t & 31;
    const int wave = t >> 5;      // 0..7
    const int wm   = wave & 3;    // M sub-group (32 rows)
    const int wn   = wave >> 2;   // N sub-group (64 cols)
    const int h    = lane >> 4;   // K-half select (0/1)
    const int r    = lane & 15;   // row/col within 16

    const int blockN0 = blockIdx.x * BN;
    const int blockM0 = blockIdx.y * BM;

    v8f acc[2][4];
#pragma unroll
    for (int tm = 0; tm < 2; ++tm)
#pragma unroll
        for (int tn = 0; tn < 4; ++tn) {
            if (ACC) {
#pragma unroll
                for (int v = 0; v < 8; ++v) {
                    int gm = blockM0 + wm * 32 + tm * 16 + v + 8 * h;
                    int gn = blockN0 + wn * 64 + tn * 16 + r;
                    acc[tm][tn][v] = C[(size_t)gm * ldc + gn];
                }
            } else {
#pragma unroll
                for (int v = 0; v < 8; ++v) acc[tm][tn][v] = 0.0f;
            }
        }

    for (int k0 = 0; k0 < K; k0 += BK) {
        __syncthreads();

        // ---- stage A tile (BM x BK) as bf16 into As[m][k] ----
        if (!TRANSA) {
            const float* Ab = A + (size_t)blockM0 * lda + k0;
#pragma unroll
            for (int i = 0; i < 4; ++i) {
                int g = t + i * 256;          // float4 group 0..1023
                int m = g >> 3;               // 0..127
                int k = (g & 7) * 4;          // 0..28
                const float4 v4 = *(const float4*)(Ab + (size_t)m * lda + k);
                unsigned* dst = (unsigned*)&As[m * LDSS + k];
                dst[0] = pack2bf(v4.x, v4.y);
                dst[1] = pack2bf(v4.z, v4.w);
            }
            if (k0 + BK < K) {
                int m = t >> 1, k = (t & 1) * 16;
                __builtin_prefetch(Ab + (size_t)m * lda + BK + k, 0, 0);
            }
        } else {
            const float* Ab = A + (size_t)k0 * lda + blockM0;
#pragma unroll
            for (int i = 0; i < 16; ++i) {
                int linear = t + i * 256;
                int m = linear & 127, kk = linear >> 7;   // coalesced along m
                As[m * LDSS + kk] = f2bf(Ab[(size_t)kk * lda + m]);
            }
            if (k0 + BK < K) {
                int kk = t >> 3, m = (t & 7) * 16;
                __builtin_prefetch(Ab + (size_t)(BK + kk) * lda + m, 0, 0);
            }
        }

        // ---- stage B tile (BK x BN) transposed into Bs[n][k] ----
        if (!TRANSB) {
            const float* Bb = B + (size_t)k0 * ldb + blockN0;
#pragma unroll
            for (int i = 0; i < 4; ++i) {
                int g = t + i * 256;          // float4 group 0..1023
                int k = g >> 5;               // 0..31
                int n = (g & 31) * 4;         // 0..124
                const float4 v4 = *(const float4*)(Bb + (size_t)k * ldb + n);
                Bs[(n + 0) * LDSS + k] = f2bf(v4.x);
                Bs[(n + 1) * LDSS + k] = f2bf(v4.y);
                Bs[(n + 2) * LDSS + k] = f2bf(v4.z);
                Bs[(n + 3) * LDSS + k] = f2bf(v4.w);
            }
            if (k0 + BK < K) {
                int k = t >> 3, n = (t & 7) * 16;
                __builtin_prefetch(Bb + (size_t)(BK + k) * ldb + n, 0, 0);
            }
        } else {
            // global B stored [n][k] (e.g. Wfc, used as Wfc^T): contiguous in k
            const float* Bb = B + (size_t)blockN0 * ldb + k0;
#pragma unroll
            for (int i = 0; i < 4; ++i) {
                int g = t + i * 256;
                int n = g >> 3;               // 0..127
                int k = (g & 7) * 4;          // 0..28
                const float4 v4 = *(const float4*)(Bb + (size_t)n * ldb + k);
                unsigned* dst = (unsigned*)&Bs[n * LDSS + k];
                dst[0] = pack2bf(v4.x, v4.y);
                dst[1] = pack2bf(v4.z, v4.w);
            }
        }
        __syncthreads();

        // ---- fragments per ISA 16-bit A(16x32)/B(32x16) lane layouts ----
        BF16x16 afrag[2], bfrag[4];
#pragma unroll
        for (int tm = 0; tm < 2; ++tm) {
            int m = wm * 32 + tm * 16 + r;
            const unsigned short* arow = &As[m * LDSS];
#pragma unroll
            for (int j = 0; j < 8; ++j) {
                afrag[tm].u[j]     = arow[h * 8 + j];        // K = h*8 + j
                afrag[tm].u[8 + j] = arow[16 + h * 8 + j];   // K = 16 + h*8 + j
            }
        }
#pragma unroll
        for (int tn = 0; tn < 4; ++tn) {
            int n = wn * 64 + tn * 16 + r;
            const unsigned short* brow = &Bs[n * LDSS];
#pragma unroll
            for (int j = 0; j < 8; ++j) {
                bfrag[tn].u[j]     = brow[h * 8 + j];        // K = h*8 + j
                bfrag[tn].u[8 + j] = brow[16 + h * 8 + j];   // K = 16 + h*8 + j
            }
        }

        // ---- 8 WMMA per wave per K-step ----
#pragma unroll
        for (int tm = 0; tm < 2; ++tm)
#pragma unroll
            for (int tn = 0; tn < 4; ++tn)
                acc[tm][tn] = __builtin_amdgcn_wmma_f32_16x16x32_bf16(
                    false, afrag[tm].v, false, bfrag[tn].v,
                    (short)0, acc[tm][tn], false, false);
    }

    // ---- store D (optionally tanh), per 16x16 C/D VGPR layout ----
#pragma unroll
    for (int tm = 0; tm < 2; ++tm)
#pragma unroll
        for (int tn = 0; tn < 4; ++tn)
#pragma unroll
            for (int v = 0; v < 8; ++v) {
                int gm = blockM0 + wm * 32 + tm * 16 + v + 8 * h;
                int gn = blockN0 + wn * 64 + tn * 16 + r;
                float val = acc[tm][tn][v];
                if (ACT) val = tanhf(val);
                D[(size_t)gm * ldc + gn] = val;
            }
}

// ---------------------------------------------------------------------------
// Elementwise / segment helpers
// ---------------------------------------------------------------------------
__global__ void zero_kernel(float* __restrict__ p, int n) {
    int i = blockIdx.x * blockDim.x + threadIdx.x;
    if (i < n) p[i] = 0.0f;
}

__global__ void seg_gather_sum_kernel(const float* __restrict__ x0,
                                      const int* __restrict__ idx,
                                      const int* __restrict__ seg,
                                      float* __restrict__ h0, int T, int O) {
    int i = blockIdx.x * blockDim.x + threadIdx.x;
    if (i >= T * O) return;
    int tt = i / O, o = i - tt * O;
    atomicAdd(&h0[(size_t)seg[tt] * O + o], x0[(size_t)idx[tt] * O + o]);
}

__global__ void seg_count_kernel(const int* __restrict__ seg, float* __restrict__ cnt, int T) {
    int i = blockIdx.x * blockDim.x + threadIdx.x;
    if (i < T) atomicAdd(&cnt[seg[i]], 1.0f);
}

__global__ void seg_mean_kernel(float* __restrict__ h0, const float* __restrict__ cnt,
                                int n, int O) {
    int i = blockIdx.x * blockDim.x + threadIdx.x;
    if (i < n) h0[i] = h0[i] / fmaxf(cnt[i / O], 1.0f);
}

// ---------------------------------------------------------------------------
// Host-side orchestration
// ---------------------------------------------------------------------------
static const int cN0 = 3072, cN1 = 6144, cN2 = 3072;
static const int cF = 128, cH = 256, cO = 128;
static const int cNH = 1024, cT = 8192;

extern "C" void kernel_launch(void* const* d_in, const int* in_sizes, int n_in,
                              void* d_out, int out_size, void* d_ws, size_t ws_size,
                              hipStream_t stream) {
    const float* X0  = (const float*)d_in[0];
    const float* X1  = (const float*)d_in[1];
    const float* X2  = (const float*)d_in[2];
    const float* L0  = (const float*)d_in[3];
    const float* L1  = (const float*)d_in[4];
    const float* L2  = (const float*)d_in[5];
    const float* Lu0 = (const float*)d_in[6];
    const float* Lu1 = (const float*)d_in[7];
    const float* Ld1 = (const float*)d_in[8];
    const float* Ld2 = (const float*)d_in[9];
    const float* B1  = (const float*)d_in[10];
    const float* B2  = (const float*)d_in[11];
    const float* W1  = (const float*)d_in[12];
    const float* W2  = (const float*)d_in[13];
    const float* W3  = (const float*)d_in[14];
    const float* Wfc = (const float*)d_in[15];
    const int* hidx  = (const int*)d_in[16];
    const int* hseg  = (const int*)d_in[17];
    float* out = (float*)d_out;

    // workspace carve-up (f32)
    float* ws = (float*)d_ws;
    size_t off = 0;
    auto alloc = [&](size_t n) { float* p = ws + off; off += n; return p; };
    float* xA0 = alloc((size_t)cN0 * cH);
    float* xA1 = alloc((size_t)cN1 * cH);
    float* xA2 = alloc((size_t)cN2 * cH);
    float* xB0 = alloc((size_t)cN0 * cH);
    float* xB1 = alloc((size_t)cN1 * cH);
    float* xB2 = alloc((size_t)cN2 * cH);
    float* o0  = alloc((size_t)cN0 * cH);
    float* o1  = alloc((size_t)cN1 * cH);
    float* o2  = alloc((size_t)cN2 * cH);
    float* tmp = alloc((size_t)cN1 * cH);
    float* h0  = alloc((size_t)cNH * cO);
    float* cnt = alloc((size_t)cNH);

    // product: C = A @ B                      (plain or transA)
    auto prod = [&](float* C, const float* A, const float* Bm,
                    int M, int N, int K, int lda, int ldb, int ldc, int tA) {
        dim3 grid(N / BN, M / BM);
        if (tA)
            gemm_wmma_bf16<1, 0, 0, 0><<<grid, 256, 0, stream>>>(A, Bm, C, C, M, N, K, lda, ldb, ldc);
        else
            gemm_wmma_bf16<0, 0, 0, 0><<<grid, 256, 0, stream>>>(A, Bm, C, C, M, N, K, lda, ldb, ldc);
    };
    // accumulate: C += A @ B ; if act: D = tanh(C + A@B)
    auto acc = [&](float* C, float* D, const float* A, const float* Bm,
                   int M, int N, int K, int lda, int ldb, int ldc, int act) {
        dim3 grid(N / BN, M / BM);
        if (act)
            gemm_wmma_bf16<0, 0, 1, 1><<<grid, 256, 0, stream>>>(A, Bm, C, D, M, N, K, lda, ldb, ldc);
        else
            gemm_wmma_bf16<0, 0, 1, 0><<<grid, 256, 0, stream>>>(A, Bm, C, C, M, N, K, lda, ldb, ldc);
    };
    // projection with B stored transposed (Wfc^T)
    auto projT = [&](float* C, const float* A, const float* Bm,
                     int M, int N, int K, int lda, int ldb, int ldc) {
        dim3 grid(N / BN, M / BM);
        gemm_wmma_bf16<0, 1, 0, 0><<<grid, 256, 0, stream>>>(A, Bm, C, C, M, N, K, lda, ldb, ldc);
    };
    auto ew = [&](int n) { return dim3((n + 255) / 256); };

    // one SCN layer (tanh fused into final accumulating GEMM of each chain)
    auto layer = [&](const float* x0, const float* x1, const float* x2,
                     int din, const float* W, int dout,
                     float* nx0, float* nx1, float* nx2) {
        const float* Wid = W;
        const float* WL  = W + 1 * (size_t)din * dout;
        const float* Wu  = W + 2 * (size_t)din * dout;
        const float* Wd  = W + 3 * (size_t)din * dout;
        const float* Wb  = W + 4 * (size_t)din * dout;
        const float* Wc  = W + 5 * (size_t)din * dout;

        // o0 = tanh(x0@Wid + (L0@x0)@WL + (Lu0@x0)@Wu + (B1@x1)@Wb)
        prod(o0, x0, Wid, cN0, dout, din, din, dout, dout, 0);
        prod(tmp, L0, x0, cN0, din, cN0, cN0, din, din, 0);
        acc(o0, o0, tmp, WL, cN0, dout, din, din, dout, dout, 0);
        prod(tmp, Lu0, x0, cN0, din, cN0, cN0, din, din, 0);
        acc(o0, o0, tmp, Wu, cN0, dout, din, din, dout, dout, 0);
        prod(tmp, B1, x1, cN0, din, cN1, cN1, din, din, 0);
        acc(o0, nx0, tmp, Wb, cN0, dout, din, din, dout, dout, 1);

        // o1 = tanh(x1@Wid + (L1@x1)@WL + (Lu1@x1)@Wu + (Ld1@x1)@Wd
        //           + (B2@x2)@Wb + (B1.T@x0)@Wc)
        prod(o1, x1, Wid, cN1, dout, din, din, dout, dout, 0);
        prod(tmp, L1, x1, cN1, din, cN1, cN1, din, din, 0);
        acc(o1, o1, tmp, WL, cN1, dout, din, din, dout, dout, 0);
        prod(tmp, Lu1, x1, cN1, din, cN1, cN1, din, din, 0);
        acc(o1, o1, tmp, Wu, cN1, dout, din, din, dout, dout, 0);
        prod(tmp, Ld1, x1, cN1, din, cN1, cN1, din, din, 0);
        acc(o1, o1, tmp, Wd, cN1, dout, din, din, dout, dout, 0);
        prod(tmp, B2, x2, cN1, din, cN2, cN2, din, din, 0);
        acc(o1, o1, tmp, Wb, cN1, dout, din, din, dout, dout, 0);
        prod(tmp, B1, x0, cN1, din, cN0, cN1, din, din, 1);       // B1.T @ x0
        acc(o1, nx1, tmp, Wc, cN1, dout, din, din, dout, dout, 1);

        // o2 = tanh(x2@Wid + (L2@x2)@WL + (Ld2@x2)@Wd + (B2.T@x1)@Wc)
        prod(o2, x2, Wid, cN2, dout, din, din, dout, dout, 0);
        prod(tmp, L2, x2, cN2, din, cN2, cN2, din, din, 0);
        acc(o2, o2, tmp, WL, cN2, dout, din, din, dout, dout, 0);
        prod(tmp, Ld2, x2, cN2, din, cN2, cN2, din, din, 0);
        acc(o2, o2, tmp, Wd, cN2, dout, din, din, dout, dout, 0);
        prod(tmp, B2, x1, cN2, din, cN1, cN2, din, din, 1);       // B2.T @ x1
        acc(o2, nx2, tmp, Wc, cN2, dout, din, din, dout, dout, 1);
    };

    // three layers: 128 -> 256 -> 256 -> 128, ping-pong A/B activation sets
    layer(X0, X1, X2, cF, W1, cH, xB0, xB1, xB2);
    layer(xB0, xB1, xB2, cH, W2, cH, xA0, xA1, xA2);
    layer(xA0, xA1, xA2, cH, W3, cO, xB0, xB1, xB2);

    // hyperedge segment mean over node features (x0 final, stride = O)
    zero_kernel<<<ew(cNH * cO), 256, 0, stream>>>(h0, cNH * cO);
    zero_kernel<<<ew(cNH), 256, 0, stream>>>(cnt, cNH);
    seg_count_kernel<<<ew(cT), 256, 0, stream>>>(hseg, cnt, cT);
    seg_gather_sum_kernel<<<ew(cT * cO), 256, 0, stream>>>(xB0, hidx, hseg, h0, cT, cO);
    seg_mean_kernel<<<ew(cNH * cO), 256, 0, stream>>>(h0, cnt, cNH * cO, cO);

    // final projections with Wfc^T, written directly into concatenated output
    projT(out,                            h0,  Wfc, cNH, cO, cO, cO, cO, cO);
    projT(out + (size_t)cNH * cO,         xB1, Wfc, cN1, cO, cO, cO, cO, cO);
    projT(out + (size_t)(cNH + cN1) * cO, xB2, Wfc, cN2, cO, cO, cO, cO, cO);
}
